// MultiHead_SelfAttention_66228395705178
// MI455X (gfx1250) — matrix operand order repaired and checked by reading, not compiled, not run on hardware
//
#include <hip/hip_runtime.h>
#include <hip/hip_bf16.h>
#include <stdint.h>

// ---------------------------------------------------------------------------
// MHA (Q=K=V per reference bug), flash-attention style, bf16 WMMA on gfx1250.
// B=2, S=2048, D=1024, H=16, Hd=64.
//
// Pipeline:
//   0) f32 -> bf16 pre-convert of x, Wq, Wo (memory-bound, L2-resident after)
//   1) qproj: q = x @ Wq^T + bq  (bf16 WMMA, pure load+wmma inner loop)
//   2) attn : flash attention, KV tiles double-buffered in LDS via
//             global_load_async_to_lds_b128 (+ s_wait_asynccnt)
//   3) oproj: out = attn @ Wo^T + bo (f32 out)
// ---------------------------------------------------------------------------

constexpr int kB    = 2;
constexpr int kSeq  = 2048;
constexpr int kDim  = 1024;
constexpr int kH    = 16;
constexpr int kHd   = 64;

typedef __attribute__((ext_vector_type(16))) __bf16 v16bf;
typedef __attribute__((ext_vector_type(8)))  __bf16 v8bf;
typedef __attribute__((ext_vector_type(8)))  float  v8f;

__device__ inline v16bf cat8(v8bf lo, v8bf hi) {
  return __builtin_shufflevector(lo, hi, 0,1,2,3,4,5,6,7,8,9,10,11,12,13,14,15);
}

// ---------------------------------------------------------------------------
// Kernel 0: bulk f32 -> bf16 conversion (8 floats / thread).
// ---------------------------------------------------------------------------
__global__ void __launch_bounds__(256)
f32_to_bf16_kernel(const float* __restrict__ src, __bf16* __restrict__ dst, int n8) {
  const int i = blockIdx.x * 256 + threadIdx.x;
  if (i >= n8) return;
  const float4* s = (const float4*)src + (size_t)i * 2;
  const float4 a = s[0], b = s[1];
  v8bf o;
  o[0]=(__bf16)a.x; o[1]=(__bf16)a.y; o[2]=(__bf16)a.z; o[3]=(__bf16)a.w;
  o[4]=(__bf16)b.x; o[5]=(__bf16)b.y; o[6]=(__bf16)b.z; o[7]=(__bf16)b.w;
  *((v8bf*)dst + i) = o;
}

// ---------------------------------------------------------------------------
// Kernel 1: q = x @ Wq^T + bq, written as bf16 in [B,H,S,Hd] layout.
// Block = 256 threads (8 waves, 2Mx4N); block tile 64x256; wave tile 32x64.
// A/B fragments loaded directly as bf16 -> inner loop is load + wmma only.
// ---------------------------------------------------------------------------
__global__ void __launch_bounds__(256)
qproj_kernel(const __bf16* __restrict__ xb, const __bf16* __restrict__ Wqb,
             const float* __restrict__ bq, __bf16* __restrict__ Qh) {
  const int lane  = threadIdx.x & 31;
  const int wave  = threadIdx.x >> 5;
  const int warpM = wave >> 2;          // 0..1
  const int warpN = wave & 3;           // 0..3
  const int m0 = blockIdx.x * 64  + warpM * 32;
  const int n0 = blockIdx.y * 256 + warpN * 64;
  const int r    = lane & 15;
  const int half = lane >> 4;

  v8f acc[2][4] = {};

  for (int k0 = 0; k0 < kDim; k0 += 32) {
    v16bf a[2];
#pragma unroll
    for (int mi = 0; mi < 2; ++mi) {
      // A 16x32 layout: half 0 -> K {0..7,16..23}, half 1 -> K {8..15,24..31}
      const __bf16* ap = xb + (size_t)(m0 + mi*16 + r) * kDim + k0 + half*8;
      a[mi] = cat8(*(const v8bf*)ap, *(const v8bf*)(ap + 16));
    }
    v16bf b[4];
#pragma unroll
    for (int ni = 0; ni < 4; ++ni) {
      // B 32x16 layout: half 0 -> K {0..7,16..23}, half 1 -> K {8..15,24..31}
      const __bf16* bp = Wqb + (size_t)(n0 + ni*16 + r) * kDim + k0 + half*8;
      b[ni] = cat8(*(const v8bf*)bp, *(const v8bf*)(bp + 16));
    }
#pragma unroll
    for (int mi = 0; mi < 2; ++mi)
#pragma unroll
      for (int ni = 0; ni < 4; ++ni)
        acc[mi][ni] = __builtin_amdgcn_wmma_f32_16x16x32_bf16(
            false, a[mi], false, b[ni], (short)0, acc[mi][ni], false, false);
  }

  // Epilogue: bias + scatter to [B,H,S,Hd] as bf16.
#pragma unroll
  for (int mi = 0; mi < 2; ++mi) {
    const int mbase = m0 + mi*16 + half*8;   // rows mbase..mbase+7
#pragma unroll
    for (int ni = 0; ni < 4; ++ni) {
      const int n  = n0 + ni*16 + r;
      const int h  = n >> 6;
      const int hd = n & 63;
      const float bias = bq[n];
#pragma unroll
      for (int i = 0; i < 8; ++i) {
        const int row = mbase + i;
        const int bb  = row >> 11;           // row / 2048
        const int s   = row & 2047;
        Qh[(((size_t)bb * kH + h) * kSeq + s) * kHd + hd] = (__bf16)(acc[mi][ni][i] + bias);
      }
    }
  }
}

// ---------------------------------------------------------------------------
// Kernel 2: flash attention per (b,h). Block = 256 thr (8 waves), 128 Q rows.
// KV tile (32 keys x 64 dims bf16) double-buffered in LDS; next tile's async
// DMA overlaps the current tile's WMMA + softmax.
// ---------------------------------------------------------------------------
__global__ void __launch_bounds__(256)
attn_kernel(const __bf16* __restrict__ Qh, __bf16* __restrict__ attnOut) {
  constexpr float kScale = 0.125f;                 // 1/sqrt(64)
  __shared__ __attribute__((aligned(16))) __bf16 kv[2][32][kHd];    // 8 KB
  __shared__ __attribute__((aligned(16))) __bf16 pt[8][16][32];     // 8 KB

  const int lane = threadIdx.x & 31;
  const int wave = threadIdx.x >> 5;
  const int r    = lane & 15;
  const int half = lane >> 4;
  const int h    = blockIdx.y;
  const int b    = blockIdx.z;
  const int q0   = blockIdx.x * 128 + wave * 16;

  const __bf16* base = Qh + (size_t)(b * kH + h) * kSeq * kHd;

  // Q fragments (16 rows x 64 dims -> two 16x32 A fragments), loaded once.
  v16bf qa[2];
  {
    const __bf16* qp = base + (size_t)(q0 + r) * kHd + half*8;
#pragma unroll
    for (int kk = 0; kk < 2; ++kk) {
      v8bf lo = *(const v8bf*)(qp + kk*32);
      v8bf hi = *(const v8bf*)(qp + kk*32 + 16);
      qa[kk] = cat8(lo, hi);
    }
  }

  float mrow[8], lrow[8];
#pragma unroll
  for (int i = 0; i < 8; ++i) { mrow[i] = -1e30f; lrow[i] = 0.0f; }
  v8f o[4] = {};

  const int ldr = threadIdx.x >> 3;        // key row 0..31
  const int ldc = (threadIdx.x & 7) * 8;   // bf16 column chunk (16B)

  // Preload tile 0 into buffer 0.
  {
    const uint64_t g = (uint64_t)(base + (size_t)ldr * kHd + ldc);
    const uint32_t l = (uint32_t)(size_t)&kv[0][ldr][ldc];
    asm volatile("global_load_async_to_lds_b128 %0, %1, off"
                 :: "v"(l), "v"(g) : "memory");
  }

  constexpr int kIters = kSeq / 32;
  for (int it = 0; it < kIters; ++it) {
    const int cur = it & 1;
    asm volatile("s_wait_asynccnt 0" ::: "memory");
    __syncthreads();   // tile[cur] filled & visible; tile[cur^1] fully consumed

    if (it + 1 < kIters) {
      const uint64_t g = (uint64_t)(base + (size_t)((it + 1) * 32 + ldr) * kHd + ldc);
      const uint32_t l = (uint32_t)(size_t)&kv[cur ^ 1][ldr][ldc];
      asm volatile("global_load_async_to_lds_b128 %0, %1, off"
                   :: "v"(l), "v"(g) : "memory");
    }

    // ---- scores: two 16-key column tiles, each = 2 WMMA over Hd=64 ----
    v8f s[2];
#pragma unroll
    for (int c = 0; c < 2; ++c) {
      const __bf16* kp = &kv[cur][c*16 + r][0];
      const int kb = half * 8;
      v16bf bkLo = cat8(*(const v8bf*)(kp + kb),      *(const v8bf*)(kp + kb + 16));
      v16bf bkHi = cat8(*(const v8bf*)(kp + 32 + kb), *(const v8bf*)(kp + 48 + kb));
      v8f z = {};
      z = __builtin_amdgcn_wmma_f32_16x16x32_bf16(false, qa[0], false, bkLo, (short)0, z, false, false);
      z = __builtin_amdgcn_wmma_f32_16x16x32_bf16(false, qa[1], false, bkHi, (short)0, z, false, false);
      s[c] = z;
    }

    // ---- online softmax (rows half*8+i, cols spread across 16 lanes) ----
    float pr0[8], pr1[8];
#pragma unroll
    for (int i = 0; i < 8; ++i) {
      float v0 = s[0][i] * kScale;
      float v1 = s[1][i] * kScale;
      float mx = fmaxf(v0, v1);
      mx = fmaxf(mx, __shfl_xor(mx, 1));
      mx = fmaxf(mx, __shfl_xor(mx, 2));
      mx = fmaxf(mx, __shfl_xor(mx, 4));
      mx = fmaxf(mx, __shfl_xor(mx, 8));
      const float mnew = fmaxf(mrow[i], mx);
      const float f = __expf(mrow[i] - mnew);
      pr0[i] = __expf(v0 - mnew);
      pr1[i] = __expf(v1 - mnew);
      float rs = pr0[i] + pr1[i];
      rs += __shfl_xor(rs, 1);
      rs += __shfl_xor(rs, 2);
      rs += __shfl_xor(rs, 4);
      rs += __shfl_xor(rs, 8);
      lrow[i] = lrow[i] * f + rs;
      mrow[i] = mnew;
#pragma unroll
      for (int j = 0; j < 4; ++j) o[j][i] *= f;
    }

    // ---- transpose P (C-layout -> A-layout) through per-wave LDS ----
#pragma unroll
    for (int i = 0; i < 8; ++i) {
      pt[wave][half*8 + i][r]      = (__bf16)pr0[i];
      pt[wave][half*8 + i][16 + r] = (__bf16)pr1[i];
    }
    asm volatile("s_wait_dscnt 0" ::: "memory");
    __builtin_amdgcn_wave_barrier();
    v16bf pa;
    {
      const __bf16* pp = &pt[wave][r][0] + half*8;
      pa = cat8(*(const v8bf*)(pp), *(const v8bf*)(pp + 16));
    }

    // ---- O += P @ V (V = same kv tile, B-layout over 32 keys) ----
#pragma unroll
    for (int j = 0; j < 4; ++j) {
      const int dcol = j*16 + r;
      const int kb   = half * 8;
      v16bf vb;
#pragma unroll
      for (int e = 0; e < 16; ++e) vb[e] = kv[cur][kb + e + (e >= 8 ? 8 : 0)][dcol];
      o[j] = __builtin_amdgcn_wmma_f32_16x16x32_bf16(
          false, pa, false, vb, (short)0, o[j], false, false);
    }
  }

  // ---- normalize and write attn output in [B,S,D] (bf16) ----
#pragma unroll
  for (int j = 0; j < 4; ++j) {
    const int dg = h * kHd + j*16 + r;
#pragma unroll
    for (int i = 0; i < 8; ++i) {
      const int srow = q0 + half*8 + i;
      attnOut[((size_t)b * kSeq + srow) * kDim + dg] = (__bf16)(o[j][i] / lrow[i]);
    }
  }
}

// ---------------------------------------------------------------------------
// Kernel 3: out = attn @ Wo^T + bo (f32 output). Same skeleton as kernel 1.
// ---------------------------------------------------------------------------
__global__ void __launch_bounds__(256)
oproj_kernel(const __bf16* __restrict__ attn, const __bf16* __restrict__ Wob,
             const float* __restrict__ bo, float* __restrict__ out) {
  const int lane  = threadIdx.x & 31;
  const int wave  = threadIdx.x >> 5;
  const int warpM = wave >> 2;
  const int warpN = wave & 3;
  const int m0 = blockIdx.x * 64  + warpM * 32;
  const int n0 = blockIdx.y * 256 + warpN * 64;
  const int r    = lane & 15;
  const int half = lane >> 4;

  v8f acc[2][4] = {};

  for (int k0 = 0; k0 < kDim; k0 += 32) {
    v16bf a[2];
#pragma unroll
    for (int mi = 0; mi < 2; ++mi) {
      const __bf16* ap = attn + (size_t)(m0 + mi*16 + r) * kDim + k0 + half*8;
      a[mi] = cat8(*(const v8bf*)ap, *(const v8bf*)(ap + 16));
    }
    v16bf b[4];
#pragma unroll
    for (int ni = 0; ni < 4; ++ni) {
      const __bf16* bp = Wob + (size_t)(n0 + ni*16 + r) * kDim + k0 + half*8;
      b[ni] = cat8(*(const v8bf*)bp, *(const v8bf*)(bp + 16));
    }
#pragma unroll
    for (int mi = 0; mi < 2; ++mi)
#pragma unroll
      for (int ni = 0; ni < 4; ++ni)
        acc[mi][ni] = __builtin_amdgcn_wmma_f32_16x16x32_bf16(
            false, a[mi], false, b[ni], (short)0, acc[mi][ni], false, false);
  }

#pragma unroll
  for (int mi = 0; mi < 2; ++mi) {
    const int mbase = m0 + mi*16 + half*8;
#pragma unroll
    for (int ni = 0; ni < 4; ++ni) {
      const int n = n0 + ni*16 + r;
      const float bias = bo[n];
#pragma unroll
      for (int i = 0; i < 8; ++i)
        out[(size_t)(mbase + i) * kDim + n] = acc[mi][ni][i] + bias;
    }
  }
}

// ---------------------------------------------------------------------------
extern "C" void kernel_launch(void* const* d_in, const int* in_sizes, int n_in,
                              void* d_out, int out_size, void* d_ws, size_t ws_size,
                              hipStream_t stream) {
  const float* x  = (const float*)d_in[0];
  const float* Wq = (const float*)d_in[1];
  const float* bq = (const float*)d_in[2];
  const float* Wo = (const float*)d_in[3];
  const float* bo = (const float*)d_in[4];
  float* out = (float*)d_out;

  // Workspace layout (bytes):
  //   [0,        8M)  Qh      bf16 [B,H,S,Hd]
  //   [8M,      16M)  attnOut bf16 [B,S,D]
  //   [16M,     24M)  xb      bf16 [B*S, D]
  //   [24M,     26M)  Wqb     bf16 [D, D]
  //   [26M,     28M)  Wob     bf16 [D, D]
  const size_t qhBytes = (size_t)kB * kH * kSeq * kHd * sizeof(__bf16);   // 8 MB
  const size_t xBytes  = (size_t)kB * kSeq * kDim * sizeof(__bf16);       // 8 MB
  const size_t wBytes  = (size_t)kDim * kDim * sizeof(__bf16);            // 2 MB
  char* ws = (char*)d_ws;
  __bf16* Qh   = (__bf16*)(ws);
  __bf16* attn = (__bf16*)(ws + qhBytes);
  __bf16* xb   = (__bf16*)(ws + qhBytes + xBytes);
  __bf16* Wqb  = (__bf16*)(ws + qhBytes + 2 * xBytes);
  __bf16* Wob  = (__bf16*)(ws + qhBytes + 2 * xBytes + wBytes);

  dim3 blk(256);

  // 0) pre-convert to bf16
  const int nx8 = kB * kSeq * kDim / 8;   // 524288
  const int nw8 = kDim * kDim / 8;        // 131072
  f32_to_bf16_kernel<<<dim3((nx8 + 255) / 256), blk, 0, stream>>>(x,  xb,  nx8);
  f32_to_bf16_kernel<<<dim3((nw8 + 255) / 256), blk, 0, stream>>>(Wq, Wqb, nw8);
  f32_to_bf16_kernel<<<dim3((nw8 + 255) / 256), blk, 0, stream>>>(Wo, Wob, nw8);

  // 1) q projection: M=4096 rows / 64, N=1024 cols / 256
  qproj_kernel<<<dim3(64, 4), blk, 0, stream>>>(xb, Wqb, bq, Qh);
  // 2) attention: (S/128, H, B)
  attn_kernel<<<dim3(kSeq / 128, kH, kB), blk, 0, stream>>>(Qh, attn);
  // 3) output projection
  oproj_kernel<<<dim3(64, 4), blk, 0, stream>>>(attn, Wob, bo, out);
}
